// ExpansionContrastModule_89644557402722
// MI455X (gfx1250) — matrix operand order repaired and checked by reading, not compiled, not run on hardware
//
#include <hip/hip_runtime.h>
#include <hip/hip_fp16.h>
#include <math.h>

typedef __attribute__((ext_vector_type(16))) _Float16 v16h;
typedef __attribute__((ext_vector_type(8)))  _Float16 v8h;
typedef __attribute__((ext_vector_type(8)))  float    v8f;

#define HID   16
#define IN_C  256
#define OUT_C 128
#define Hh    192
#define Ww    192
#define Bb    4
#define NPIX  (Bb*Hh*Ww)

// workspace offsets in _Float16 units (all fragment regions are contiguous:
// 206 fragments of 512 halves each, then the h tensor)
#define OFF_W0 0                      // 8 frags  : in_w0  (16x256), chunks of K=32
#define OFF_W1 (OFF_W0 + 8*512)       // 72 frags : in_w1  (16x2304) tap-major
#define OFF_WM (OFF_W1 + 72*512)      // 54 frags : w1/w2/w3 per branch/group, sign-folded, K padded 16->32
#define OFF_OW (OFF_WM + 54*512)      // 72 frags : out_w (128x288) m-tile major
#define OFF_H  (OFF_OW + 72*512)      // h: [branch][pixel][16ch] f16, 2*NPIX*16 halves

// ---------------------------------------------------------------------------
// Kernel 0: repack all weights into WMMA A-fragment order (f16).
// A 16x32 f16 fragment, packed as [lane][elem]:
//   m = lane & 15 ;  K = ((lane&16)?16:0) + e
// ---------------------------------------------------------------------------
__global__ __launch_bounds__(256) void pack_kernel(
    const float* __restrict__ in_w0, const float* __restrict__ in_w1,
    const float* __restrict__ w10, const float* __restrict__ w20, const float* __restrict__ w30,
    const float* __restrict__ w11, const float* __restrict__ w21, const float* __restrict__ w31,
    const float* __restrict__ out_w, _Float16* __restrict__ ws) {
  int tid = blockIdx.x * blockDim.x + threadIdx.x;
  if (tid >= 206 * 512) return;
  int frag = tid >> 9;
  int idx  = tid & 511;
  int l = idx >> 4, e = idx & 15;
  int m = l & 15;
  int K = ((l & 16) ? 16 : 0) + e;
  float v = 0.f;
  if (frag < 8) {                                   // in_w0: (16,256,1,1)
    v = in_w0[m * IN_C + frag * 32 + K];
  } else if (frag < 80) {                           // in_w1: (16,256,3,3), frag=(tap*8+kc)
    int t = (frag - 8) >> 3, kc = (frag - 8) & 7;
    int ky = t / 3, kx = t % 3;
    int c = kc * 32 + K;
    v = in_w1[((m * IN_C + c) * 3 + ky) * 3 + kx];
  } else if (frag < 134) {                          // w{1,2,3}_{0,1}: (9,16,16), K padded, sign folded
    int q = frag - 80;
    int br = q / 27, r = q % 27, mat = r / 9, g = r % 9;
    if (K < 16) {
      const float* w = br ? (mat == 0 ? w11 : mat == 1 ? w21 : w31)
                          : (mat == 0 ? w10 : mat == 1 ? w20 : w30);
      v = w[(g * HID + m) * HID + K] * (g < 8 ? -1.f : 1.f);
    }
  } else {                                          // out_w: (128,288,1,1), frag=(mt*9+kc)
    int q = frag - 134;
    int mt = q / 9, kc = q % 9;
    v = out_w[(mt * 16 + m) * 288 + kc * 32 + K];
  }
  ws[tid] = (_Float16)v;
}

// ---------------------------------------------------------------------------
// Kernel 1: input projections h0 (1x1 conv) and h1 (3x3 conv) via implicit
// GEMM. Block = 128 threads (4 waves), covers (b, y, 64 px). LDS stages a
// 32-channel chunk of cen, channel-major so B fragments are contiguous.
// ---------------------------------------------------------------------------
__global__ __launch_bounds__(128) void proj_kernel(
    const float* __restrict__ cen,
    const float* __restrict__ b0, const float* __restrict__ b1,
    _Float16* __restrict__ ws) {
  __shared__ __align__(32) _Float16 tile[3 * 66 * 32];   // [row][col][chan]
  int tid  = threadIdx.x;
  int lane = tid & 31, wave = tid >> 5;
  int x0 = blockIdx.x * 64;
  int y  = blockIdx.y;
  int bb = blockIdx.z;
  int n    = lane & 15;
  int off8 = (lane & 16) ? 8 : 0;
  int kb_b = (lane & 16) ? 16 : 0;
  int pxb  = wave * 16;

  v8f acc0 = {}, acc1 = {};
  for (int kc = 0; kc < 8; ++kc) {
    __syncthreads();
    for (int idx = tid; idx < 3 * 66 * 32; idx += 128) {
      int col = idx % 66;
      int r2  = idx / 66;
      int row = r2 % 3;
      int c   = r2 / 3;
      int yy = y + row - 1;
      int xx = x0 - 1 + col;
      float v = 0.f;
      if (yy >= 0 && yy < Hh && xx >= 0 && xx < Ww)
        v = cen[(((size_t)bb * IN_C + kc * 32 + c) * Hh + yy) * Ww + xx];
      tile[(row * 66 + col) * 32 + c] = (_Float16)v;
    }
    __syncthreads();
    v16h a0 = *(const v16h*)&ws[OFF_W0 + kc * 512 + lane * 16];
#pragma unroll
    for (int t = 0; t < 9; ++t) {
      int ky = t / 3, kx = t % 3;
      v16h bf = *(const v16h*)&tile[(ky * 66 + pxb + n + kx) * 32 + kb_b];
      v16h a1 = *(const v16h*)&ws[OFF_W1 + (t * 8 + kc) * 512 + lane * 16];
      acc1 = __builtin_amdgcn_wmma_f32_16x16x32_f16(false, a1, false, bf, (short)0, acc1, false, false);
      if (t == 4)
        acc0 = __builtin_amdgcn_wmma_f32_16x16x32_f16(false, a0, false, bf, (short)0, acc0, false, false);
    }
  }
  size_t p = ((size_t)bb * Hh + y) * Ww + (x0 + pxb + n);
  v8h h0, h1;
#pragma unroll
  for (int e = 0; e < 8; ++e) {
    h0[e] = (_Float16)(acc0[e] + b0[off8 + e]);
    h1[e] = (_Float16)(acc1[e] + b1[off8 + e]);
  }
  *(v8h*)&ws[OFF_H + p * 16 + off8] = h0;
  *(v8h*)&ws[OFF_H + (size_t)NPIX * 16 + p * 16 + off8] = h1;
}

// ---------------------------------------------------------------------------
// Kernel 2: per wave (16 px): o1/o2/o3 group matmuls (K=16 padded to 32) via
// WMMA from shifted h reads, per-pixel 9x9 softmax attention (VALU), build
// 288-channel cat in LDS, then fused final 128x288 GEMM via WMMA.
// Block = 64 threads (2 waves), covers (b, y, 32 px).
// ---------------------------------------------------------------------------
__global__ __launch_bounds__(64) void attn_kernel(
    const float* __restrict__ scale, const float* __restrict__ out_b,
    const _Float16* __restrict__ ws, float* __restrict__ out) {
  __shared__ __align__(32) _Float16 s_om[2 * 3 * 9 * 16 * 16];  // [wave][mat][g][pix][c]
  __shared__ __align__(32) _Float16 s_cat[2 * 16 * 288];        // [wave][pix][ch]
  int tid  = threadIdx.x;
  int lane = tid & 31, wave = tid >> 5;
  int x0 = blockIdx.x * 32 + wave * 16;
  int y  = blockIdx.y;
  int bb = blockIdx.z;
  int n    = lane & 15;
  int off8 = (lane & 16) ? 8 : 0;
  int kb_b = (lane & 16) ? 16 : 0;
  _Float16* om  = &s_om[wave * 3 * 9 * 16 * 16];
  _Float16* cat = &s_cat[wave * 16 * 288];

  for (int br = 0; br < 2; ++br) {
    int s = br ? 5 : 1;
    const int dy[9] = {-s, -s, -s, 0, s, s, s, 0, 0};
    const int dx[9] = {-s, 0, s, s, s, 0, -s, -s, 0};
    const _Float16* hb = ws + OFF_H + (size_t)br * NPIX * 16;
#pragma unroll
    for (int g = 0; g < 9; ++g) {
      int yy = y + dy[g];
      int xx = x0 + n + dx[g];
      v16h bf = {};  // lanes >=16 carry the K=16..31 zero padding
      if (lane < 16 && yy >= 0 && yy < Hh && xx >= 0 && xx < Ww)
        bf = *(const v16h*)&hb[(((size_t)bb * Hh + yy) * Ww + xx) * 16];
#pragma unroll
      for (int mat = 0; mat < 3; ++mat) {
        v16h af = *(const v16h*)&ws[OFF_WM + (size_t)((br * 3 + mat) * 9 + g) * 512 + lane * 16];
        v8f z = {};
        v8f o = __builtin_amdgcn_wmma_f32_16x16x32_f16(false, af, false, bf, (short)0, z, false, false);
        v8h oh;
#pragma unroll
        for (int e = 0; e < 8; ++e) oh[e] = (_Float16)o[e];
        *(v8h*)&om[((mat * 9 + g) * 16 + n) * 16 + off8] = oh;
      }
    }
    __syncthreads();
    // per-pixel 9x9 attention: lanes 0-15 handle p=0..4 of pixel n, 16-31 p=5..8
    float scv = scale[br];
    int pix = lane & 15;
    int p0 = (lane < 16) ? 0 : 5;
    int p1 = (lane < 16) ? 5 : 9;
    for (int p = p0; p < p1; ++p) {
      float o2p[16];
#pragma unroll
      for (int c = 0; c < 16; ++c)
        o2p[c] = (float)om[((1 * 9 + p) * 16 + pix) * 16 + c];
      float lg[9], mx = -1e30f;
#pragma unroll
      for (int q = 0; q < 9; ++q) {
        const _Float16* o1q = &om[((0 * 9 + q) * 16 + pix) * 16];
        float d = 0.f;
#pragma unroll
        for (int c = 0; c < 16; ++c) d += o2p[c] * (float)o1q[c];
        lg[q] = scv * d;
        mx = fmaxf(mx, lg[q]);
      }
      float se = 0.f;
#pragma unroll
      for (int q = 0; q < 9; ++q) { lg[q] = __expf(lg[q] - mx); se += lg[q]; }
      float inv = 1.f / se;
      float oacc[16];
#pragma unroll
      for (int c = 0; c < 16; ++c) oacc[c] = 0.f;
#pragma unroll
      for (int q = 0; q < 9; ++q) {
        float a = lg[q] * inv;
        const _Float16* o3q = &om[((2 * 9 + q) * 16 + pix) * 16];
#pragma unroll
        for (int c = 0; c < 16; ++c) oacc[c] += a * (float)o3q[c];
      }
      v16h ov;
#pragma unroll
      for (int c = 0; c < 16; ++c) ov[c] = (_Float16)oacc[c];
      *(v16h*)&cat[pix * 288 + br * 144 + p * 16] = ov;
    }
    __syncthreads();  // cat complete; omats safe to overwrite next branch
  }
  // fused final 1x1 conv: out(128x288) @ cat(288 x 16 px)
#pragma unroll
  for (int mt = 0; mt < 8; ++mt) {
    v8f acc = {};
#pragma unroll
    for (int kc = 0; kc < 9; ++kc) {
      v16h af = *(const v16h*)&ws[OFF_OW + (size_t)(mt * 9 + kc) * 512 + lane * 16];
      v16h bf = *(const v16h*)&cat[n * 288 + kc * 32 + kb_b];
      acc = __builtin_amdgcn_wmma_f32_16x16x32_f16(false, af, false, bf, (short)0, acc, false, false);
    }
#pragma unroll
    for (int e = 0; e < 8; ++e) {
      int o = mt * 16 + off8 + e;
      out[(((size_t)bb * OUT_C + o) * Hh + y) * Ww + (x0 + n)] = acc[e] + out_b[o];
    }
  }
}

// ---------------------------------------------------------------------------
extern "C" void kernel_launch(void* const* d_in, const int* in_sizes, int n_in,
                              void* d_out, int out_size, void* d_ws, size_t ws_size,
                              hipStream_t stream) {
  const float* cen   = (const float*)d_in[0];
  const float* in_w0 = (const float*)d_in[1];
  const float* in_b0 = (const float*)d_in[2];
  const float* in_w1 = (const float*)d_in[3];
  const float* in_b1 = (const float*)d_in[4];
  const float* w10   = (const float*)d_in[5];
  const float* w20   = (const float*)d_in[6];
  const float* w30   = (const float*)d_in[7];
  const float* w11   = (const float*)d_in[8];
  const float* w21   = (const float*)d_in[9];
  const float* w31   = (const float*)d_in[10];
  const float* scale = (const float*)d_in[11];
  const float* out_w = (const float*)d_in[12];
  const float* out_b = (const float*)d_in[13];
  _Float16* ws = (_Float16*)d_ws;
  float* out = (float*)d_out;

  pack_kernel<<<(206 * 512 + 255) / 256, 256, 0, stream>>>(
      in_w0, in_w1, w10, w20, w30, w11, w21, w31, out_w, ws);
  proj_kernel<<<dim3(Ww / 64, Hh, Bb), 128, 0, stream>>>(cen, in_b0, in_b1, ws);
  attn_kernel<<<dim3(Ww / 32, Hh, Bb), 64, 0, stream>>>(scale, out_b, ws, out);
}
